// ESN_65197603554238
// MI455X (gfx1250) — matrix operand order repaired and checked by reading, not compiled, not run on hardware
//
#include <hip/hip_runtime.h>
#include <hip/hip_bf16.h>
#include <math.h>

// ---------------------------------------------------------------------------
// ESN (leaky echo-state network) for MI455X / gfx1250.
// B=8, T=1024, D=128, U=2048, leak=0.5, square even-indexed neurons.
// ---------------------------------------------------------------------------

#define B_   8
#define T_   1024
#define D_   128
#define U_   2048
#define OUTF (D_ + U_)          // 2176
#define NBLK 32                 // persistent scan workgroups (one per WGP)
#define NSL  (U_ / NBLK)        // 64 reservoir columns per scan workgroup

typedef __attribute__((ext_vector_type(16))) __bf16 bf16x16;
typedef __attribute__((ext_vector_type(8)))  float  f32x8;
typedef __attribute__((ext_vector_type(4)))  int    v4i;

union AFrag { uint4 u[2]; bf16x16 v; };
union CFrag { f32x8 v; float f[8]; };

// ---- workspace layout (bytes) ---------------------------------------------
#define OFF_PROJ  ((size_t)0)                                   // f32 [8192][2048]
#define OFF_WREST (OFF_PROJ  + (size_t)B_ * T_ * U_ * 4)        // bf16 [U][U]   (w_res^T)
#define OFF_INB   (OFF_WREST + (size_t)U_ * U_ * 2)             // bf16 [8192][128]
#define OFF_WINT  (OFF_INB   + (size_t)B_ * T_ * D_ * 2)        // bf16 [U][D]   (w_in^T)
#define OFF_STATE (OFF_WINT  + (size_t)U_ * D_ * 2)             // bf16 [2][8][2048]
#define OFF_CNT   (OFF_STATE + (size_t)2 * B_ * U_ * 2)         // u32  [2048] barrier ctrs
#define ZERO_DW   ((int)((2 * B_ * U_ * 2 + 2048 * 4) / 4))     // dwords: state + counters

// ---- CDNA5 async global->LDS copy (ASYNCcnt path), with safe fallback -----
#if defined(__HIP_DEVICE_COMPILE__) && defined(__gfx1250__) && \
    __has_builtin(__builtin_amdgcn_global_load_async_to_lds_b128)
#define USE_ASYNC_LDS 1
#else
#define USE_ASYNC_LDS 0
#endif

__device__ __forceinline__ void copy16_g2l(const uint4* g, uint4* l) {
#if USE_ASYNC_LDS
    __builtin_amdgcn_global_load_async_to_lds_b128(
        (__attribute__((address_space(1))) v4i*)g,
        (__attribute__((address_space(3))) v4i*)l,
        0, 0);
#else
    *l = *g;   // global load -> VGPR -> ds_store fallback
#endif
}

__device__ __forceinline__ void async_copies_wait() {
#if USE_ASYNC_LDS
#if __has_builtin(__builtin_amdgcn_s_wait_asynccnt)
    __builtin_amdgcn_s_wait_asynccnt(0);
#else
    asm volatile("s_wait_asynccnt 0" ::: "memory");
#endif
#endif
}

__device__ __forceinline__ unsigned short f2bf(float f) {
    unsigned int u = __float_as_uint(f);
    // round-to-nearest-even bf16
    unsigned int r = (u + 0x7FFFu + ((u >> 16) & 1u)) >> 16;
    return (unsigned short)r;
}

// ---------------------------------------------------------------------------
__global__ void zero_ws(unsigned int* p, int n) {
    int i = blockIdx.x * blockDim.x + threadIdx.x;
    if (i < n) p[i] = 0u;
}

// inputs fp32 -> bf16, and write the concat(inputs, .) part of the output.
__global__ void conv_inputs(const float* __restrict__ in,
                            unsigned short* __restrict__ inb,
                            float* __restrict__ out) {
    int i = blockIdx.x * blockDim.x + threadIdx.x;   // < 8192*128
    float v = in[i];
    inb[i] = f2bf(v);
    int r = i >> 7, d = i & 127;                     // r = b*T + t
    out[(size_t)r * OUTF + d] = v;
}

// w_in [D][U] fp32 -> w_in^T [U][D] bf16
__global__ void conv_win(const float* __restrict__ w_in,
                         unsigned short* __restrict__ winT) {
    int i = blockIdx.x * blockDim.x + threadIdx.x;   // < U*D
    int u = i >> 7, d = i & 127;
    winT[i] = f2bf(w_in[(size_t)d * U_ + u]);
}

// w_res [K][N] fp32 -> w_res^T [N][K] bf16
__global__ void conv_wres(const float* __restrict__ w_res,
                          unsigned short* __restrict__ wrT) {
    int i = blockIdx.x * blockDim.x + threadIdx.x;   // < U*U
    int n = i >> 11, k = i & 2047;
    wrT[i] = f2bf(w_res[(size_t)k * U_ + n]);
}

// ---------------------------------------------------------------------------
// proj[r][n] = sum_d inputs_bf16[r][d] * w_in^T[n][d] + b_in[n]
// grid = (8192/16, 2048/64), 128 threads = 4 waves; wave w owns one 16x16 tile.
__global__ void proj_gemm(const unsigned short* __restrict__ inb,
                          const unsigned short* __restrict__ winT,
                          const float* __restrict__ b_in,
                          float* __restrict__ proj) {
    const int wv   = threadIdx.x >> 5;
    const int lane = threadIdx.x & 31;
    const int m    = lane & 15;
    const int half = lane >> 4;
    const int r0   = blockIdx.x * 16;
    const int n0   = blockIdx.y * 64 + wv * 16;

    CFrag acc; acc.v = (f32x8){0.f,0.f,0.f,0.f,0.f,0.f,0.f,0.f};
    #pragma unroll
    for (int kb = 0; kb < D_; kb += 32) {
        AFrag a, b;
        const uint4* pa = (const uint4*)(inb  + (size_t)(r0 + m) * D_ + kb + half * 8);
        a.u[0] = pa[0]; a.u[1] = pa[2];   // K = kb+8h+{0..7}, kb+16+8h+{0..7}
        const uint4* pb = (const uint4*)(winT + (size_t)(n0 + m) * D_ + kb + half * 8);
        b.u[0] = pb[0]; b.u[1] = pb[2];   // K = kb+8h+{0..7}, kb+16+8h+{0..7}
        acc.v = __builtin_amdgcn_wmma_f32_16x16x32_bf16(
            false, a.v, false, b.v, (short)0, acc.v, false, false);
    }
    const int n = n0 + m;
    const float bias = b_in[n];
    #pragma unroll
    for (int v = 0; v < 8; ++v) {
        int row = r0 + v + half * 8;      // C layout: lanes 0-15 -> M=v, 16-31 -> M=v+8
        proj[(size_t)row * U_ + n] = acc.f[v] + bias;
    }
}

// ---------------------------------------------------------------------------
// Persistent scan: 32 workgroups x 128 threads. Each WG pins its 64-column
// slice of w_res^T (256 KB bf16) in LDS for all 1024 steps; state (8x2048
// bf16) is staged into LDS each step (async global->LDS on gfx1250); one
// device-wide barrier per step.
__global__ void __launch_bounds__(128, 1)
esn_scan(const unsigned short* __restrict__ wrT,
         const float* __restrict__ proj,
         float* __restrict__ out,
         unsigned short* __restrict__ state,     // [2][8][2048] bf16, buf0 zeroed
         unsigned int* __restrict__ cnt) {       // [T] zeroed barrier counters
    extern __shared__ char smem[];
    unsigned short* ldsB = (unsigned short*)smem;                       // [64][2048]
    unsigned short* ldsA = (unsigned short*)(smem + (size_t)NSL * U_ * 2); // [8][2048]

    const int tid  = threadIdx.x;
    const int wv   = tid >> 5;
    const int lane = tid & 31;
    const int m    = lane & 15;
    const int half = lane >> 4;
    const int blk  = blockIdx.x;
    const int ncol = blk * NSL + wv * 16 + m;    // valid for lanes 0-15

    // One-time preload of this WG's w_res^T slice into LDS (256 KB).
    {
        const uint4* src = (const uint4*)(wrT + (size_t)blk * NSL * U_);
        uint4* dst = (uint4*)ldsB;
        const int nvec = NSL * U_ / 8;           // 16384 uint4
        for (int i = tid; i < nvec; i += 128) copy16_g2l(src + i, dst + i);
    }

    float sp[B_];
    #pragma unroll
    for (int i = 0; i < B_; ++i) sp[i] = 0.f;

    for (int t = 0; t < T_; ++t) {
        // Stage the broadcast state (read buffer t&1) into LDS.
        {
            const uint4* sa = (const uint4*)(state + (size_t)(t & 1) * B_ * U_);
            uint4* da = (uint4*)ldsA;
            #pragma unroll
            for (int i = 0; i < B_ * U_ / 8 / 128; ++i)   // 16 x b128 per thread
                copy16_g2l(sa + tid + i * 128, da + tid + i * 128);
        }
        async_copies_wait();
        __syncthreads();

        CFrag acc; acc.v = (f32x8){0.f,0.f,0.f,0.f,0.f,0.f,0.f,0.f};
        for (int kb = 0; kb < U_; kb += 32) {
            AFrag a, b;
            if (m < B_) {                         // rows 8..15 of A are zero padding
                const uint4* pa = (const uint4*)(ldsA + (size_t)m * U_ + kb + half * 8);
                a.u[0] = pa[0]; a.u[1] = pa[2];
            } else {
                uint4 z = {0u, 0u, 0u, 0u};
                a.u[0] = z; a.u[1] = z;
            }
            const uint4* pb = (const uint4*)(ldsB + (size_t)(wv * 16 + m) * U_ + kb + half * 8);
            b.u[0] = pb[0]; b.u[1] = pb[2];
            acc.v = __builtin_amdgcn_wmma_f32_16x16x32_bf16(
                false, a.v, false, b.v, (short)0, acc.v, false, false);
        }

        unsigned short* st_out = state + (size_t)((t & 1) ^ 1) * B_ * U_;
        if (lane < 16) {                          // lanes 0-15 hold rows M=0..7
            #pragma unroll
            for (int bq = 0; bq < B_; ++bq) {
                float pre = acc.f[bq] + proj[(size_t)(bq * T_ + t) * U_ + ncol];
                float nw  = 0.5f * sp[bq] + 0.5f * tanhf(pre);
                sp[bq] = nw;
                st_out[(size_t)bq * U_ + ncol] = f2bf(nw);
                float pw = ((ncol & 1) == 0) ? nw * nw : nw;  // PowerIndex
                out[(size_t)(bq * T_ + t) * OUTF + D_ + ncol] = pw;
            }
        }
        __syncthreads();                          // all waves done with ldsA

        if (t + 1 < T_) {                         // device-wide step barrier
            __threadfence();
            if (tid == 0)
                __hip_atomic_fetch_add(&cnt[t], 1u, __ATOMIC_RELEASE,
                                       __HIP_MEMORY_SCOPE_AGENT);
            while (__hip_atomic_load(&cnt[t], __ATOMIC_ACQUIRE,
                                     __HIP_MEMORY_SCOPE_AGENT) < (unsigned)NBLK) {
#if defined(__HIP_DEVICE_COMPILE__)
                __builtin_amdgcn_s_sleep(1);
#endif
            }
        }
    }
}

// ---------------------------------------------------------------------------
extern "C" void kernel_launch(void* const* d_in, const int* in_sizes, int n_in,
                              void* d_out, int out_size, void* d_ws, size_t ws_size,
                              hipStream_t stream) {
    const float* inputs = (const float*)d_in[0];   // [8,1024,128]
    const float* w_in   = (const float*)d_in[1];   // [128,2048]
    const float* b_in   = (const float*)d_in[2];   // [2048]
    const float* w_res  = (const float*)d_in[3];   // [2048,2048]
    float* out = (float*)d_out;                    // [8,1024,2176]
    char*  ws  = (char*)d_ws;

    float*          proj  = (float*)(ws + OFF_PROJ);
    unsigned short* wrT   = (unsigned short*)(ws + OFF_WREST);
    unsigned short* inb   = (unsigned short*)(ws + OFF_INB);
    unsigned short* winT  = (unsigned short*)(ws + OFF_WINT);
    unsigned short* state = (unsigned short*)(ws + OFF_STATE);
    unsigned int*   cnt   = (unsigned int*)(ws + OFF_CNT);

    (void)in_sizes; (void)n_in; (void)out_size; (void)ws_size;

    // Zero state double-buffer + barrier counters (deterministic per launch).
    zero_ws<<<(ZERO_DW + 255) / 256, 256, 0, stream>>>(
        (unsigned int*)(ws + OFF_STATE), ZERO_DW);

    conv_inputs<<<(B_ * T_ * D_) / 256, 256, 0, stream>>>(inputs, inb, out);
    conv_win  <<<(U_ * D_) / 256, 256, 0, stream>>>(w_in, winT);
    conv_wres <<<(U_ * U_) / 256, 256, 0, stream>>>(w_res, wrT);

    proj_gemm<<<dim3(B_ * T_ / 16, U_ / 64), 128, 0, stream>>>(inb, winT, b_in, proj);

    const size_t lds_bytes = ((size_t)NSL * U_ + (size_t)B_ * U_) * 2; // 288 KB
    esn_scan<<<NBLK, 128, lds_bytes, stream>>>(wrT, proj, out, state, cnt);
}
